// MultiHeadLatentAttention_43319040147649
// MI455X (gfx1250) — compile-verified, matrix operand order checked by reading, not run on hardware
//
#include <hip/hip_runtime.h>

// ---------------------------------------------------------------------------
// MLA forward for MI455X (gfx1250, wave32, v_wmma_f32_16x16x32_f16).
//
// Compute-bound (~212 GFLOP vs ~130MB unique bytes, L2-resident at 23.3TB/s).
// f16 operands everywhere (one-time cvt of x + transposed weights), WMMA with
// f32 accumulation for all matmuls. GEMM tiles are staged by the CDNA5
// Tensor Data Mover (TENSOR_LOAD_TO_LDS, TENSORcnt) — one DMA per tile with
// hardware LDS padding and hardware OOB zero-fill — into double-buffered
// LDS; per-lane GLOBAL_LOAD_ASYNC_TO_LDS_B128 is the fallback path.
// ---------------------------------------------------------------------------

typedef __attribute__((ext_vector_type(16))) _Float16     v16h;
typedef __attribute__((ext_vector_type(8)))  float        v8f;
typedef __attribute__((ext_vector_type(4)))  _Float16     v4h;
typedef __attribute__((ext_vector_type(4)))  unsigned int u32x4;
typedef __attribute__((ext_vector_type(4)))  int          i32x4;
typedef __attribute__((ext_vector_type(8)))  int          i32x8;

#define kB     2
#define kS     2048
#define kD     2048
#define kH     16
#define kNOPE  128
#define kROPE  64
#define kVD    128
#define kKVR   512
#define kQR    1536
#define kT     (kB * kS)        // 4096 tokens
#define kHD    (kNOPE + kROPE)  // 192
#define kEPS   1e-6f

#define HAVE_TDM __has_builtin(__builtin_amdgcn_tensor_load_to_lds)

// ---------------------------------------------------------------------------
// CDNA5 async global->LDS copy (16B per lane), fallback staging path.
// ---------------------------------------------------------------------------
__device__ __forceinline__ void async_copy_b128(const _Float16* gsrc, void* ldst) {
  const unsigned int loff = (unsigned int)(size_t)ldst;  // LDS offset = low 32b
  asm volatile("global_load_async_to_lds_b128 %0, %1, off"
               :: "v"(loff), "v"(gsrc) : "memory");
}
__device__ __forceinline__ void wait_async0() {
#if __has_builtin(__builtin_amdgcn_s_wait_asynccnt)
  __builtin_amdgcn_s_wait_asynccnt(0);
#else
  asm volatile("s_wait_asynccnt 0" ::: "memory");
#endif
}
__device__ __forceinline__ void wait_tensor0() {
#if __has_builtin(__builtin_amdgcn_s_wait_tensorcnt)
  __builtin_amdgcn_s_wait_tensorcnt(0);
#else
  asm volatile("s_wait_tensorcnt 0" ::: "memory");
#endif
}

#if HAVE_TDM
// ---------------------------------------------------------------------------
// TDM: load a (rows x 32-half) f16 tile from global (row stride srcStrideHalfs)
// into LDS with hardware row padding 32->40 halfs (pad_interval=3: 16 DWORDs,
// pad_amount=3: 4 DWORDs). Rows >= validRows are hardware zero-filled
// (tensor_dim1 OOB). D# layout per ISA 8.3-8.4; groups 2/3 zero (2D tensor).
// ---------------------------------------------------------------------------
__device__ __forceinline__ void tdm_load_tile(const _Float16* gsrc, void* ldst,
                                              int rows, int validRows,
                                              int srcStrideHalfs) {
  const unsigned long long ga = (unsigned long long)(size_t)gsrc;
  u32x4 g0;
  g0[0] = 1u;                                          // count=1 (valid, user)
  g0[1] = (unsigned int)(size_t)ldst;                  // lds_addr
  g0[2] = (unsigned int)(ga & 0xffffffffu);            // global_addr[31:0]
  g0[3] = (unsigned int)((ga >> 32) & 0x01ffffffu)     // global_addr[56:32]
        | (2u << 30);                                  // type=2 ("image")
  i32x8 g1;
  g1[0] = (1 << 16)                                    // data_size=1 (2 bytes)
        | (1 << 20)                                    // pad_enable
        | (3 << 22)                                    // pad_interval: 16 DW
        | (3 << 25);                                   // pad_amount: 4 DW
  g1[1] = (32 << 16);                                  // tensor_dim0[15:0]=32
  g1[2] = (validRows & 0xffff) << 16;                  // tensor_dim1[15:0]
  g1[3] = ((validRows >> 16) & 0xffff) | (32 << 16);   // dim1[31:16] | tile_dim0=32
  g1[4] = rows & 0xffff;                               // tile_dim1 (tile_dim2=0)
  g1[5] = srcStrideHalfs;                              // tensor_dim0_stride[31:0]
  g1[6] = 0;                                           // stride0[47:32] | stride1[15:0]
  g1[7] = 0;                                           // tensor_dim1_stride[47:16]
  i32x4 gz = {0, 0, 0, 0};
#if defined(__clang_major__) && (__clang_major__ >= 23)
  i32x8 z8 = {0, 0, 0, 0, 0, 0, 0, 0};
  __builtin_amdgcn_tensor_load_to_lds(g0, g1, gz, gz, z8, 0);
#else
  __builtin_amdgcn_tensor_load_to_lds(g0, g1, gz, gz, 0);
#endif
}
#endif  // HAVE_TDM

// ---------------------------------------------------------------------------
// WMMA fragment helpers (ISA 7.12.2, wave32).
// ---------------------------------------------------------------------------
__device__ __forceinline__ v16h load_frag(const _Float16* __restrict__ base, int ldk) {
  const int lane = threadIdx.x & 31;
  const _Float16* p = base + (lane & 15) * ldk + ((lane >> 4) << 3);
  v16h r;
#pragma unroll
  for (int j = 0; j < 16; ++j) {
    const int v = j >> 1;
    const int k = ((v >> 2) << 4) + ((v & 3) << 1) + (j & 1);
    r[j] = p[k];
  }
  return r;
}

__device__ __forceinline__ v8f wmma32(v16h a, v16h b, v8f c) {
  return __builtin_amdgcn_wmma_f32_16x16x32_f16(false, a, false, b,
                                                (short)0, c, false, false);
}

__device__ __forceinline__ float grp16_max(float v) {
  v = fmaxf(v, __shfl_xor(v, 1, 32));
  v = fmaxf(v, __shfl_xor(v, 2, 32));
  v = fmaxf(v, __shfl_xor(v, 4, 32));
  v = fmaxf(v, __shfl_xor(v, 8, 32));
  return v;
}
__device__ __forceinline__ float grp16_sum(float v) {
  v += __shfl_xor(v, 1, 32);
  v += __shfl_xor(v, 2, 32);
  v += __shfl_xor(v, 4, 32);
  v += __shfl_xor(v, 8, 32);
  return v;
}

// ---------------------------------------------------------------------------
// One-time converters.
// ---------------------------------------------------------------------------
__global__ __launch_bounds__(256) void cvt_f16_kernel(
    const float* __restrict__ in, _Float16* __restrict__ out, int n4)
{
  const int i = blockIdx.x * 256 + threadIdx.x;
  if (i < n4) {
    const float4 v = *(const float4*)(in + (size_t)i * 4);
    v4h h;
    h[0] = (_Float16)v.x; h[1] = (_Float16)v.y;
    h[2] = (_Float16)v.z; h[3] = (_Float16)v.w;
    *(v4h*)(out + (size_t)i * 4) = h;
  }
}

// out[n][k] = (f16) in[k][n]   (in: KxN f32 row-major -> out: NxK f16)
__global__ __launch_bounds__(256) void cvt_transpose_f16_kernel(
    const float* __restrict__ in, _Float16* __restrict__ out, int K, int N)
{
  const size_t idx = (size_t)blockIdx.x * 256 + threadIdx.x;  // n*K + k, k fast
  if (idx < (size_t)K * N) {
    const int k = (int)(idx % K);
    const int n = (int)(idx / K);
    out[idx] = (_Float16)in[(size_t)k * N + n];
  }
}

// ---------------------------------------------------------------------------
// GEMM: C(MxN) = A(MxK,f16,row) * Bt(NxK,f16,row)^T, f32 accum via WMMA.
// Block 256 thr (8 waves, 4x2), block tile 256x128x32, wave tile 64x64
// (16 WMMA per 8 fragment loads per k-step). Double-buffered LDS; TDM DMA
// (1 descriptor per tile, wave 0 issues, TENSORcnt) streams tile kt+1 while
// tile kt computes; per-lane async-DMA fallback when TDM builtin is absent.
// ---------------------------------------------------------------------------
template <typename TC, bool GUARD_N>
__global__ __launch_bounds__(256) void gemm_h_kernel(
    const _Float16* __restrict__ A, const _Float16* __restrict__ Bt,
    TC* __restrict__ C, int M, int N, int K)
{
  __shared__ _Float16 As[2][256][40];
  __shared__ _Float16 Bs[2][128][40];

  const int tid  = threadIdx.x;
  const int lane = tid & 31;
  const int wave = tid >> 5;
  const int wm   = (wave >> 1) * 64;     // 0..192
  const int wn   = (wave & 1) * 64;      // 0,64
  const int bm   = blockIdx.y * 256;
  const int bn   = blockIdx.x * 128;

  const int rr = tid >> 2;               // row group, +i*64
  const int c0 = (tid & 3) << 3;         // 8-half column chunk

#if HAVE_TDM
  const bool issuer = (tid < 32);        // wave 0 drives the TDM
  auto issueTiles = [&](int kt, int buf) {
    if (issuer) {
      tdm_load_tile(A + (size_t)bm * K + (size_t)kt * 32, &As[buf][0][0],
                    256, 256, K);
      int validB = 128;
      if (GUARD_N) { const int rem = N - bn; validB = rem < 128 ? rem : 128; }
      tdm_load_tile(Bt + (size_t)bn * K + (size_t)kt * 32, &Bs[buf][0][0],
                    128, validB, K);
    }
  };
  auto waitTiles = [&]() { if (issuer) wait_tensor0(); };
#else
  auto issueTiles = [&](int kt, int buf) {
#pragma unroll
    for (int i = 0; i < 8; ++i) {        // A: 256x32 halfs
      const int r = rr + i * 64;
      async_copy_b128(A + (size_t)(bm + r) * K + kt * 32 + c0, &As[buf][r][c0]);
    }
#pragma unroll
    for (int i = 0; i < 2; ++i) {        // Bt: 128x32 halfs
      const int r = rr + i * 64;
      if (GUARD_N && (bn + r) >= N)
        *(uint4*)&Bs[buf][r][c0] = make_uint4(0u, 0u, 0u, 0u);
      else
        async_copy_b128(Bt + (size_t)(bn + r) * K + kt * 32 + c0, &Bs[buf][r][c0]);
    }
  };
  auto waitTiles = [&]() { wait_async0(); };
#endif

  v8f acc[4][4];
#pragma unroll
  for (int i = 0; i < 4; ++i)
#pragma unroll
    for (int j = 0; j < 4; ++j) acc[i][j] = (v8f){0,0,0,0,0,0,0,0};

  const int ntiles = K >> 5;
  issueTiles(0, 0);
  int cur = 0;

  for (int kt = 0; kt < ntiles; ++kt) {
    waitTiles();                          // tile-kt DMA landed (issuing wave)
    __syncthreads();                      // visible to all; prev reads done
    if (kt + 1 < ntiles) issueTiles(kt + 1, cur ^ 1);   // stream next tile
    if (kt + 2 < ntiles) {                // L2 warm-up two tiles ahead
      __builtin_prefetch(A  + (size_t)(bm + rr) * K + (kt + 2) * 32 + c0, 0, 3);
      __builtin_prefetch(Bt + (size_t)(bn + rr) * K + (kt + 2) * 32 + c0, 0, 3);
    }

    v16h af[4], bf[4];
#pragma unroll
    for (int i = 0; i < 4; ++i) af[i] = load_frag(&As[cur][wm + i * 16][0], 40);
#pragma unroll
    for (int j = 0; j < 4; ++j) bf[j] = load_frag(&Bs[cur][wn + j * 16][0], 40);
#pragma unroll
    for (int i = 0; i < 4; ++i)
#pragma unroll
      for (int j = 0; j < 4; ++j) acc[i][j] = wmma32(af[i], bf[j], acc[i][j]);

    cur ^= 1;
  }

  const int n_lane = lane & 15;
  const int m_hi   = (lane >> 4) * 8;
#pragma unroll
  for (int i = 0; i < 4; ++i)
#pragma unroll
    for (int j = 0; j < 4; ++j) {
      const int gm = bm + wm + i * 16 + m_hi;
      const int gn = bn + wn + j * 16 + n_lane;
      if (!GUARD_N || gn < N) {
#pragma unroll
        for (int r = 0; r < 8; ++r)
          C[(size_t)(gm + r) * N + gn] = (TC)acc[i][j][r];
      }
    }
}

// ---------------------------------------------------------------------------
// RMSNorm: block per row; f32 in -> f16 out.
// ---------------------------------------------------------------------------
__global__ __launch_bounds__(256) void rmsnorm_kernel(
    const float* __restrict__ in, const float* __restrict__ w,
    _Float16* __restrict__ out, int W, int in_stride, int out_stride)
{
  const int row = blockIdx.x;
  const float* src = in + (size_t)row * in_stride;
  _Float16* dst = out + (size_t)row * out_stride;

  float vals[8];
  float acc = 0.f;
  int cnt = 0;
  for (int i = threadIdx.x; i < W; i += 256) {
    const float v = src[i];
    vals[cnt++] = v;
    acc += v * v;
  }
#pragma unroll
  for (int m = 1; m < 32; m <<= 1) acc += __shfl_xor(acc, m, 32);
  __shared__ float red[8];
  if ((threadIdx.x & 31) == 0) red[threadIdx.x >> 5] = acc;
  __syncthreads();
  float tot = 0.f;
#pragma unroll
  for (int i = 0; i < 8; ++i) tot += red[i];
  const float rs = rsqrtf(tot / (float)W + kEPS);
  cnt = 0;
  for (int i = threadIdx.x; i < W; i += 256)
    dst[i] = (_Float16)(vals[cnt++] * rs * w[i]);
}

// ---------------------------------------------------------------------------
// RoPE + 1/sqrt(192) scale on q (f16 in-place). grid (T, H), block 128.
// ---------------------------------------------------------------------------
__global__ __launch_bounds__(128) void rope_scale_q_kernel(_Float16* __restrict__ q)
{
  const int t = blockIdx.x, h = blockIdx.y;
  const int pos = t & (kS - 1);
  _Float16* qh = q + (size_t)t * (kH * kHD) + h * kHD;
  const float scale = 0.07216878364870323f;  // 1/sqrt(192)
  const int i = threadIdx.x;
  if (i < 32) {
    const float inv = __powf(10000.f, -2.f * (float)i / 64.f);
    float sn, cs;
    __sincosf((float)pos * inv, &sn, &cs);
    const float x1 = (float)qh[kNOPE + i];
    const float x2 = (float)qh[kNOPE + 32 + i];
    qh[kNOPE + i]      = (_Float16)((x1 * cs - x2 * sn) * scale);
    qh[kNOPE + 32 + i] = (_Float16)((x2 * cs + x1 * sn) * scale);
  }
  qh[i] = (_Float16)((float)qh[i] * scale);
}

// ---------------------------------------------------------------------------
// RoPE on k_pe: ckv f32 [T][576] (pe at 512:576) -> kpe f16 [T][64].
// ---------------------------------------------------------------------------
__global__ __launch_bounds__(32) void rope_k_kernel(
    const float* __restrict__ ckv, _Float16* __restrict__ kpe)
{
  const int t = blockIdx.x, i = threadIdx.x;
  const int pos = t & (kS - 1);
  const float* src = ckv + (size_t)t * (kKVR + kROPE) + kKVR;
  const float inv = __powf(10000.f, -2.f * (float)i / 64.f);
  float sn, cs;
  __sincosf((float)pos * inv, &sn, &cs);
  const float x1 = src[i], x2 = src[32 + i];
  kpe[(size_t)t * kROPE + i]      = (_Float16)(x1 * cs - x2 * sn);
  kpe[(size_t)t * kROPE + 32 + i] = (_Float16)(x2 * cs + x1 * sn);
}

// ---------------------------------------------------------------------------
// Causal flash attention, all WMMA, all-f16 operands.
//  q [T][H*192] f16 (scaled+roped) | kv [T][H*256] f16 (k_nope|v)
//  kpe [T][64] f16 | out [T][H*128] f16
// Block 128 thr (4 waves) x 64 queries; key tiles of 64; online softmax.
// ---------------------------------------------------------------------------
#define QPAD 200   // 192 + 8  (400B row stride, 16B aligned)
#define VPAD 72    // 64 + 8

__global__ __launch_bounds__(128) void mla_attn_kernel(
    const _Float16* __restrict__ q, const _Float16* __restrict__ kv,
    const _Float16* __restrict__ kpe, _Float16* __restrict__ out)
{
  __shared__ _Float16 Qs[64][QPAD];
  __shared__ _Float16 Ks[64][QPAD];
  __shared__ _Float16 Vs[128][VPAD];      // [d][key] transposed
  __shared__ _Float16 Ps[4][16][VPAD];

  const int tid    = threadIdx.x;
  const int lane   = tid & 31;
  const int wave   = tid >> 5;
  const int n_lane = lane & 15;
  const int m_hi   = (lane >> 4) * 8;

  const int qt = blockIdx.x;
  const int h  = blockIdx.y;
  const int b  = blockIdx.z;
  const int q0   = qt * 64;
  const int tok0 = b * kS + q0;

  // ---- stage Q tile (64 x 192 halfs = 1536 uint4): batched loads ----
  {
    uint4 tmp[12];
#pragma unroll
    for (int i = 0; i < 12; ++i) {
      const int f = tid + i * 128;
      const int r = f / 24, c = (f % 24) * 8;
      tmp[i] = *(const uint4*)(q + (size_t)(tok0 + r) * (kH * kHD) + h * kHD + c);
    }
#pragma unroll
    for (int i = 0; i < 12; ++i) {
      const int f = tid + i * 128;
      const int r = f / 24, c = (f % 24) * 8;
      *(uint4*)&Qs[r][c] = tmp[i];
    }
  }
  __syncthreads();

  v16h qf[6];
#pragma unroll
  for (int ks = 0; ks < 6; ++ks) qf[ks] = load_frag(&Qs[wave * 16][ks * 32], QPAD);

  v8f o[8];
  float mrow[8], lrow[8];
#pragma unroll
  for (int r = 0; r < 8; ++r) { mrow[r] = -3.0e38f; lrow[r] = 0.f; }
#pragma unroll
  for (int d = 0; d < 8; ++d) o[d] = (v8f){0,0,0,0,0,0,0,0};

  const int myq    = q0 + wave * 16;
  const int ktiles = qt + 1;

  for (int kt = 0; kt < ktiles; ++kt) {
    const int k0    = kt * 64;
    const int ktok0 = b * kS + k0;

    {
      uint4 kn[8], kp[4], vv[8];
#pragma unroll
      for (int i = 0; i < 8; ++i) {     // k_nope 64x128 = 1024 uint4
        const int f = tid + i * 128;
        const int r = f >> 4, c = (f & 15) << 3;
        kn[i] = *(const uint4*)(kv + (size_t)(ktok0 + r) * (kH * 256) + h * 256 + c);
      }
#pragma unroll
      for (int i = 0; i < 4; ++i) {     // k_pe 64x64 = 512 uint4
        const int f = tid + i * 128;
        const int r = f >> 3, c = (f & 7) << 3;
        kp[i] = *(const uint4*)(kpe + (size_t)(ktok0 + r) * kROPE + c);
      }
#pragma unroll
      for (int i = 0; i < 8; ++i) {     // v 64x128 = 1024 uint4
        const int f = tid + i * 128;
        const int r = f >> 4, c = (f & 15) << 3;
        vv[i] = *(const uint4*)(kv + (size_t)(ktok0 + r) * (kH * 256) + h * 256 + 128 + c);
      }
#pragma unroll
      for (int i = 0; i < 8; ++i) {
        const int f = tid + i * 128;
        const int r = f >> 4, c = (f & 15) << 3;
        *(uint4*)&Ks[r][c] = kn[i];
      }
#pragma unroll
      for (int i = 0; i < 4; ++i) {
        const int f = tid + i * 128;
        const int r = f >> 3, c = (f & 7) << 3;
        *(uint4*)&Ks[r][128 + c] = kp[i];
      }
#pragma unroll
      for (int i = 0; i < 8; ++i) {     // transpose V -> Vs[d][key]
        const int f = tid + i * 128;
        const int r = f >> 4, c = (f & 15) << 3;
        union { uint4 u; _Float16 hh[8]; } tu; tu.u = vv[i];
#pragma unroll
        for (int j = 0; j < 8; ++j) Vs[c + j][r] = tu.hh[j];
      }
    }
    __syncthreads();

    // ---- S = Q K^T ----
    v8f s[4];
#pragma unroll
    for (int sub = 0; sub < 4; ++sub) s[sub] = (v8f){0,0,0,0,0,0,0,0};
#pragma unroll
    for (int sub = 0; sub < 4; ++sub)
#pragma unroll
      for (int ks = 0; ks < 6; ++ks) {
        const v16h bf = load_frag(&Ks[sub * 16][ks * 32], QPAD);
        s[sub] = wmma32(qf[ks], bf, s[sub]);
      }

    // ---- causal mask + online softmax ----
#pragma unroll
    for (int r = 0; r < 8; ++r) {
      const int qrow = myq + r + m_hi;
      float pm = -3.0e38f;
#pragma unroll
      for (int sub = 0; sub < 4; ++sub) {
        const int key = k0 + sub * 16 + n_lane;
        float sv = s[sub][r];
        if (key > qrow) sv = -1.0e30f;
        s[sub][r] = sv;
        pm = fmaxf(pm, sv);
      }
      pm = grp16_max(pm);
      const float mnew  = fmaxf(mrow[r], pm);
      const float alpha = __expf(mrow[r] - mnew);
      float ps = 0.f;
#pragma unroll
      for (int sub = 0; sub < 4; ++sub) {
        const float p = __expf(s[sub][r] - mnew);
        s[sub][r] = p;
        ps += p;
      }
      lrow[r] = lrow[r] * alpha + grp16_sum(ps);
      mrow[r] = mnew;
#pragma unroll
      for (int d = 0; d < 8; ++d) o[d][r] *= alpha;
    }

    // ---- relayout P via LDS into A-fragments ----
#pragma unroll
    for (int sub = 0; sub < 4; ++sub)
#pragma unroll
      for (int r = 0; r < 8; ++r)
        Ps[wave][r + m_hi][sub * 16 + n_lane] = (_Float16)s[sub][r];
    __syncthreads();

    const v16h pf0 = load_frag(&Ps[wave][0][0],  VPAD);
    const v16h pf1 = load_frag(&Ps[wave][0][32], VPAD);

    // ---- O += P V ----
#pragma unroll
    for (int d = 0; d < 8; ++d) {
      const v16h v0 = load_frag(&Vs[d * 16][0],  VPAD);
      const v16h v1 = load_frag(&Vs[d * 16][32], VPAD);
      o[d] = wmma32(pf0, v0, o[d]);
      o[d] = wmma32(pf1, v1, o[d]);
    }
    __syncthreads();
  }

  // ---- epilogue: O /= l -> f16 out ----
#pragma unroll
  for (int d = 0; d < 8; ++d)
#pragma unroll
    for (int r = 0; r < 8; ++r) {
      const int token = tok0 + wave * 16 + r + m_hi;
      out[(size_t)token * (kH * kVD) + h * kVD + d * 16 + n_lane] =
          (_Float16)(o[d][r] / lrow[r]);
    }
}

// ---------------------------------------------------------------------------
// Host-side pipeline
// ---------------------------------------------------------------------------
extern "C" void kernel_launch(void* const* d_in, const int* in_sizes, int n_in,
                              void* d_out, int out_size, void* d_ws, size_t ws_size,
                              hipStream_t stream)
{
  const float* x         = (const float*)d_in[0];
  const float* w_dq      = (const float*)d_in[1];
  const float* q_a_norm  = (const float*)d_in[2];
  const float* w_uq      = (const float*)d_in[3];
  const float* w_dkv     = (const float*)d_in[4];
  const float* kv_a_norm = (const float*)d_in[5];
  const float* w_ukv     = (const float*)d_in[6];
  const float* w_o       = (const float*)d_in[7];
  float* out = (float*)d_out;

  // workspace carve-up
  char* wsp = (char*)d_ws;
  auto alloc = [&](size_t bytes) {
    char* p = wsp;
    wsp += (bytes + 255) & ~(size_t)255;
    return p;
  };
  _Float16* xh     = (_Float16*)alloc((size_t)kT * kD * 2);
  _Float16* wdqT   = (_Float16*)alloc((size_t)kD * kQR * 2);          // [1536][2048]
  _Float16* wuqT   = (_Float16*)alloc((size_t)kQR * kH * kHD * 2);    // [3072][1536]
  _Float16* wdkvT  = (_Float16*)alloc((size_t)kD * (kKVR+kROPE) * 2); // [576][2048]
  _Float16* wukvT  = (_Float16*)alloc((size_t)kKVR * kH * 256 * 2);   // [4096][512]
  _Float16* woT    = (_Float16*)alloc((size_t)kD * kD * 2);           // [2048][2048]
  float*    qlat   = (float*)   alloc((size_t)kT * kQR * 4);
  _Float16* qlath  = (_Float16*)alloc((size_t)kT * kQR * 2);
  _Float16* qh     = (_Float16*)alloc((size_t)kT * kH * kHD * 2);
  float*    ckv    = (float*)   alloc((size_t)kT * (kKVR+kROPE) * 4);
  _Float16* ckvnh  = (_Float16*)alloc((size_t)kT * kKVR * 2);
  _Float16* kvh    = (_Float16*)alloc((size_t)kT * kH * 256 * 2);
  _Float16* kpeh   = (_Float16*)alloc((size_t)kT * kROPE * 2);
  _Float16* attnh  = (_Float16*)alloc((size_t)kT * kD * 2);

  dim3 blk256(256), blk128(128);
  auto cvtN = [&](const float* in, _Float16* o2, size_t n) {
    cvt_f16_kernel<<<dim3((unsigned)((n / 4 + 255) / 256)), blk256, 0, stream>>>(
        in, o2, (int)(n / 4));
  };
  auto cvtT = [&](const float* in, _Float16* o2, int K, int N) {
    const size_t n = (size_t)K * N;
    cvt_transpose_f16_kernel<<<dim3((unsigned)((n + 255) / 256)), blk256, 0, stream>>>(
        in, o2, K, N);
  };

  // 0) one-time conversions
  cvtN(x, xh, (size_t)kT * kD);
  cvtT(w_dq,  wdqT,  kD,  kQR);
  cvtT(w_uq,  wuqT,  kQR, kH * kHD);
  cvtT(w_dkv, wdkvT, kD,  kKVR + kROPE);
  cvtT(w_ukv, wukvT, kKVR, kH * 256);
  cvtT(w_o,   woT,   kD,  kD);

  // 1) q_lat = x @ w_dq                       -> f32
  gemm_h_kernel<float, false><<<dim3(kQR / 128, kT / 256), blk256, 0, stream>>>(
      xh, wdqT, qlat, kT, kQR, kD);
  // 2) rmsnorm(q_lat)                         -> f16
  rmsnorm_kernel<<<kT, blk256, 0, stream>>>(qlat, q_a_norm, qlath, kQR, kQR, kQR);
  // 3) q = q_lat_n @ w_uq                     -> f16
  gemm_h_kernel<_Float16, false><<<dim3((kH * kHD) / 128, kT / 256), blk256, 0, stream>>>(
      qlath, wuqT, qh, kT, kH * kHD, kQR);
  // 4) ckv = x @ w_dkv (N=576, TDM zero-fills OOB rows) -> f32
  gemm_h_kernel<float, true><<<dim3((kKVR + kROPE + 127) / 128, kT / 256), blk256, 0, stream>>>(
      xh, wdkvT, ckv, kT, kKVR + kROPE, kD);
  // 5) rmsnorm(c_kv)                          -> f16 packed 512-wide
  rmsnorm_kernel<<<kT, blk256, 0, stream>>>(ckv, kv_a_norm, ckvnh, kKVR,
                                            kKVR + kROPE, kKVR);
  // 6) kv = ckv_n @ w_ukv                     -> f16
  gemm_h_kernel<_Float16, false><<<dim3((kH * 256) / 128, kT / 256), blk256, 0, stream>>>(
      ckvnh, wukvT, kvh, kT, kH * 256, kKVR);
  // 7) rope + scale on q (f16 in place)
  rope_scale_q_kernel<<<dim3(kT, kH), blk128, 0, stream>>>(qh);
  // 8) rope on k_pe -> f16
  rope_k_kernel<<<kT, dim3(32), 0, stream>>>(ckv, kpeh);
  // 9) causal flash attention                 -> f16
  mla_attn_kernel<<<dim3(kS / 64, kH, kB), blk128, 0, stream>>>(qh, kvh, kpeh, attnh);
  // 10) out = attn @ w_o                      -> f32 (d_out)
  gemm_h_kernel<float, false><<<dim3(kD / 128, kT / 256), blk256, 0, stream>>>(
      attnh, woT, out, kT, kD, kD);
}
